// KANForecaster_64355789963711
// MI455X (gfx1250) — matrix operand order repaired and checked by reading, not compiled, not run on hardware
//
#include <hip/hip_runtime.h>
#include <cstdint>

// ---------------------------------------------------------------------------
// KAN forecaster on gfx1250 (MI455X).
//   layer(x) = GEMM( features(x) [B x 6I] , W [6I x O] ),
//   features = [silu(x), B0..B4(x)] per scalar; W folds scale_base /
//   scale_sp*coef.
// bf16 WMMA (v_wmma_f32_16x16x32_bf16, f32 accum):
//   L1: 32768 x 1008 x 32, L2: 32768 x 192 x 24  (~2.4 GFLOP)
//   vs ~25 MB HBM traffic (~1.1 us @ 23.3 TB/s). f32 VALU would be ~15 us
//   compute-bound; the bf16 matrix path returns this to the memory roof.
// Weight tiles staged LDS-side via the Tensor Data Mover (TDM) so weight
// staging is fully async behind the spline VALU work.
// ---------------------------------------------------------------------------

typedef __attribute__((ext_vector_type(16))) __bf16    v16bf;
typedef __attribute__((ext_vector_type(8)))  float     v8f;
typedef __attribute__((ext_vector_type(4)))  unsigned  v4u;
typedef __attribute__((ext_vector_type(8)))  int       v8i;
typedef __attribute__((ext_vector_type(4)))  int       v4i;

#if __has_builtin(__builtin_amdgcn_tensor_load_to_lds)
#define HAVE_TDM 1
#endif

__device__ __forceinline__ unsigned pack_bf16x2(float a, float b) {
    unsigned ua = __builtin_bit_cast(unsigned, a);
    unsigned ub = __builtin_bit_cast(unsigned, b);
    ua = (ua + 0x7FFFu + ((ua >> 16) & 1u)) >> 16;   // round-to-nearest-even
    ub = (ub + 0x7FFFu + ((ub >> 16) & 1u)) >> 16;
    return (ua & 0xFFFFu) | (ub << 16);
}

// Swizzle within-chunk bf16-pair index (0..47) into A-fragment order
// (ISA 7.12.2: lane<16 holds K{0-7,16-23}, lane>=16 holds K{8-15,24-31}).
__device__ __forceinline__ int swzA(int p) {
    int ks   = p >> 4;
    int q    = p & 15;
    int half = (q >> 2) & 1;
    int slot = (q & 3) | (((q >> 3) & 1) << 2);
    return ks * 16 + half * 8 + slot;
}

// ---------------------------------------------------------------------------
// Fold scales into bf16-packed weights, already padded to the 52-dword LDS
// stride: dst[chunk][o(32)][52], entries j<48 are k-pairs (B-fragment natural
// order), j>=48 are zero. TDM then copies a chunk LDS-image verbatim.
// ---------------------------------------------------------------------------
template <int I, int OUT, int NCH>
__global__ __launch_bounds__(256) void prep_w_kernel(
    const float* __restrict__ sb, const float* __restrict__ sp,
    const float* __restrict__ coef, unsigned* __restrict__ dst) {
    int idx = blockIdx.x * 256 + threadIdx.x;
    if (idx >= NCH * 32 * 52) return;
    int c   = idx / (32 * 52);
    int rem = idx - c * (32 * 52);
    int o   = rem / 52;
    int j   = rem - o * 52;
    float v[2] = {0.f, 0.f};
    if (j < 48) {
        int P = c * 48 + j;             // global bf16 pair -> k = 2P, 2P+1
#pragma unroll
        for (int t = 0; t < 2; ++t) {
            int k = 2 * P + t;
            int i = k / 6, f = k % 6;
            if (i < I && o < OUT)
                v[t] = (f == 0) ? sb[o * I + i]
                                : sp[o * I + i] * coef[(o * I + i) * 5 + (f - 1)];
        }
    }
    dst[idx] = pack_bf16x2(v[0], v[1]);
}

// ---------------------------------------------------------------------------
// Fused feature-expansion + bf16 WMMA GEMM layer.
// 256 threads (8 waves), 128 rows/block, K chunks of 16 input cols (96 K).
// Each wave: one 16-row M tile x two 16-col N tiles.
// ---------------------------------------------------------------------------
template <int NCOLS, int NCHUNKS, int OUT>
__global__ __launch_bounds__(256) void kan_layer_kernel(
    const float* __restrict__ xin, const unsigned* __restrict__ wg,
    float* __restrict__ out) {
    constexpr int MB = 128, CI = 16, FSTR = 52;     // 52-dword stride: conflict-free
    constexpr int WTILE = 32 * FSTR;                // weight chunk dwords (1664)
    // Single shared buffer so the weight tile's LDS byte offset is known.
    __shared__ __align__(16) unsigned smem[MB * FSTR + WTILE];
    unsigned*       feat  = smem;                   // A tile (swizzled pairs)
    unsigned*       wlds  = smem + MB * FSTR;       // B tile (natural pairs)
    constexpr unsigned WLDS_BYTE_OFF = MB * FSTR * 4u;

    const int tid     = threadIdx.x;
    const int lane    = tid & 31;
    const int wv      = tid >> 5;
    const int rowBase = blockIdx.x * MB;
    const int colLo   = lane & 15;
    const int halfAdd = (lane >> 4) << 3;

    v8f acc0, acc1;
#pragma unroll
    for (int r = 0; r < 8; ++r) { acc0[r] = 0.f; acc1[r] = 0.f; }

    for (int c = 0; c < NCHUNKS; ++c) {
        __syncthreads();   // previous chunk's fragment reads done

        // --- stage weight chunk (async via TDM; wave 0 issues, EXEC ignored) ---
#ifdef HAVE_TDM
        if (wv == 0) {
            unsigned long long ga =
                (unsigned long long)reinterpret_cast<uintptr_t>(wg + (size_t)c * WTILE);
            // D# group0: count=1 | lds_addr | global_addr | type=2
            v4u g0 = { 1u, WLDS_BYTE_OFF, (unsigned)ga,
                       (unsigned)((ga >> 32) & 0x1FFFFFFull) | (2u << 30) };
            // D# group1: data_size=4B; tensor_dim0=tile_dim0=1664; dim1=1;
            //            tensor_dim0_stride=1664; no mask/pad/iterate.
            v8i g1 = { (int)(2u << 16),
                       (int)((unsigned)(WTILE & 0xFFFF) << 16),
                       (int)(1u << 16),
                       (int)((unsigned)(WTILE & 0xFFFF) << 16),
                       1, (int)WTILE, 0, 0 };
            v4i gz = { 0, 0, 0, 0 };
#if __clang_major__ >= 23
            v8i gz8 = { 0, 0, 0, 0, 0, 0, 0, 0 };
            __builtin_amdgcn_tensor_load_to_lds(g0, g1, gz, gz, gz8, 0);
#else
            __builtin_amdgcn_tensor_load_to_lds(g0, g1, gz, gz, 0);
#endif
        }
#else
        for (int idx = tid; idx < WTILE / 4; idx += 256)
            reinterpret_cast<uint4*>(wlds)[idx] =
                reinterpret_cast<const uint4*>(wg + (size_t)c * WTILE)[idx];
#endif

        // --- features: 128 rows x 16 cols; float4 x loads; swizzled stores ---
        for (int e = tid; e < MB * 4; e += 256) {
            int r = e >> 2, g = e & 3;
            int col0 = c * CI + g * 4;
            bool valid = (col0 < NCOLS);
            float xv[4] = {0.f, 0.f, 0.f, 0.f};
            if (valid) {
                const float4 v4 = *reinterpret_cast<const float4*>(
                    &xin[(size_t)(rowBase + r) * NCOLS + col0]);
                xv[0] = v4.x; xv[1] = v4.y; xv[2] = v4.z; xv[3] = v4.w;
                if (c + 1 < NCHUNKS && col0 + CI < NCOLS)   // global_prefetch_b8
                    __builtin_prefetch(&xin[(size_t)(rowBase + r) * NCOLS + col0 + CI], 0, 1);
            }
            int base = r * FSTR;
#pragma unroll
            for (int u = 0; u < 4; ++u) {
                int cc = g * 4 + u;
                float f6[6] = {0.f, 0.f, 0.f, 0.f, 0.f, 0.f};
                if (valid) {
                    float x = xv[u];
                    // Cox-de Boor, uniform knots m-4 (m=0..8), spacing 1
                    float bb[8];
#pragma unroll
                    for (int m = 0; m < 8; ++m) {
                        float gl = (float)(m - 4);
                        bb[m] = (x >= gl && x < gl + 1.0f) ? 1.0f : 0.0f;
                    }
#pragma unroll
                    for (int p = 1; p <= 3; ++p) {
                        float inv = 1.0f / (float)p;
#pragma unroll
                        for (int m = 0; m < 8 - p; ++m) {
                            float gl = (float)(m - 4);       // grid[m]
                            float gr = (float)(m + p - 3);   // grid[m+p+1]
                            bb[m] = ((x - gl) * bb[m] + (gr - x) * bb[m + 1]) * inv;
                        }
                    }
                    f6[0] = x / (1.0f + __expf(-x));         // silu
#pragma unroll
                    for (int j = 0; j < 5; ++j) f6[1 + j] = bb[j];
                }
                feat[base + swzA(3 * cc + 0)] = pack_bf16x2(f6[0], f6[1]);
                feat[base + swzA(3 * cc + 1)] = pack_bf16x2(f6[2], f6[3]);
                feat[base + swzA(3 * cc + 2)] = pack_bf16x2(f6[4], f6[5]);
            }
        }

#ifdef HAVE_TDM
        if (wv == 0) __builtin_amdgcn_s_wait_tensorcnt(0);
#endif
        __syncthreads();

        // --- 3 K-steps x 2 N-tiles of v_wmma_f32_16x16x32_bf16 ---
        const int arow = wv * 16 + colLo;
#pragma unroll
        for (int ks = 0; ks < 3; ++ks) {
            union { uint4 u[2]; v16bf v; } fa, fb0, fb1;
            const uint4* ap =
                reinterpret_cast<const uint4*>(&feat[arow * FSTR + ks * 16 + halfAdd]);
            fa.u[0] = ap[0]; fa.u[1] = ap[1];
            const uint4* bp0 =
                reinterpret_cast<const uint4*>(&wlds[colLo * FSTR + ks * 16 + (halfAdd >> 1)]);
            fb0.u[0] = bp0[0]; fb0.u[1] = bp0[2];
            const uint4* bp1 =
                reinterpret_cast<const uint4*>(&wlds[(16 + colLo) * FSTR + ks * 16 + (halfAdd >> 1)]);
            fb1.u[0] = bp1[0]; fb1.u[1] = bp1[2];

            acc0 = __builtin_amdgcn_wmma_f32_16x16x32_bf16(
                false, fa.v, false, fb0.v, (short)0, acc0, false, false);
            acc1 = __builtin_amdgcn_wmma_f32_16x16x32_bf16(
                false, fa.v, false, fb1.v, (short)0, acc1, false, false);
        }
    }

    // C/D layout: lane<16 -> M = vgpr, lane>=16 -> M = 8+vgpr; N = lane&15
    const int mrow = rowBase + wv * 16 + halfAdd;
#pragma unroll
    for (int r = 0; r < 8; ++r) {
        int row = mrow + r;
        out[(size_t)row * OUT + colLo] = acc0[r];
        if (16 + colLo < OUT) out[(size_t)row * OUT + 16 + colLo] = acc1[r];
    }
}

// ---------------------------------------------------------------------------
extern "C" void kernel_launch(void* const* d_in, const int* in_sizes, int n_in,
                              void* d_out, int out_size, void* d_ws, size_t ws_size,
                              hipStream_t stream) {
    (void)in_sizes; (void)n_in; (void)out_size; (void)ws_size;
    constexpr int B = 32768, WINDOW = 168, HIDDEN = 32, HORIZON = 24;
    constexpr int NCH1 = 11, NCH2 = 2;   // ceil(168/16), 32/16
    constexpr int WTILE = 32 * 52;

    const float* x   = (const float*)d_in[0];
    const float* c1  = (const float*)d_in[1];
    const float* sb1 = (const float*)d_in[2];
    const float* sp1 = (const float*)d_in[3];
    const float* c2  = (const float*)d_in[4];
    const float* sb2 = (const float*)d_in[5];
    const float* sp2 = (const float*)d_in[6];
    float* y = (float*)d_out;

    // workspace: h [B x HIDDEN] f32, then LDS-image bf16 weight chunks
    float*    h  = (float*)d_ws;
    unsigned* w1 = (unsigned*)((char*)d_ws + (size_t)B * HIDDEN * 4);
    unsigned* w2 = w1 + NCH1 * WTILE;

    prep_w_kernel<WINDOW, HIDDEN, NCH1>
        <<<(NCH1 * WTILE + 255) / 256, 256, 0, stream>>>(sb1, sp1, c1, w1);
    prep_w_kernel<HIDDEN, HORIZON, NCH2>
        <<<(NCH2 * WTILE + 255) / 256, 256, 0, stream>>>(sb2, sp2, c2, w2);

    kan_layer_kernel<WINDOW, NCH1, HIDDEN><<<B / 128, 256, 0, stream>>>(x, w1, h);
    kan_layer_kernel<HIDDEN, NCH2, HORIZON><<<B / 128, 256, 0, stream>>>(h, w2, y);
}
